// GCN_61478161875059
// MI455X (gfx1250) — compile-verified, matrix operand order checked by reading, not run on hardware
//
#include <hip/hip_runtime.h>
#include <stdint.h>

// Problem constants (match reference)
#define NN 8192
#define DD 512
#define TWOD 1024
#define NLAYER 2
#define NEG_SLOPE 0.01f

typedef __bf16 bf16_t;
typedef __attribute__((ext_vector_type(8)))  bf16_t        v8bf;
typedef __attribute__((ext_vector_type(16))) bf16_t        v16bf;
typedef __attribute__((ext_vector_type(8)))  float         v8f;
typedef __attribute__((ext_vector_type(4)))  unsigned int  v4u;

union BFrag { v16bf v; v8bf h[2]; };

// fp32 -> bf16 (RNE) and back, via bit ops (no dependence on __bf16 arithmetic)
__device__ __forceinline__ unsigned short f2bf(float f) {
  unsigned u = __builtin_bit_cast(unsigned, f);
  unsigned r = u + 0x7FFFu + ((u >> 16) & 1u);
  return (unsigned short)(r >> 16);
}
__device__ __forceinline__ float bf2f(unsigned short h) {
  return __builtin_bit_cast(float, ((unsigned)h) << 16);
}

// Load one 16x32 A-fragment (or mirrored B-fragment) row-chunk for this lane:
// 8 bf16 at p (K 0..7 of this lane's half) and 8 bf16 at p+16 (K 16..23 half).
// p must be 16B aligned.
__device__ __forceinline__ v16bf load_frag(const unsigned short* p) {
  BFrag u;
  u.h[0] = __builtin_bit_cast(v8bf, *(const v4u*)(p));
  u.h[1] = __builtin_bit_cast(v8bf, *(const v4u*)(p + 16));
  return u.v;
}

#define WMMA_BF16(A, B, C) \
  __builtin_amdgcn_wmma_f32_16x16x32_bf16(false, (A), false, (B), (short)0, (C), false, false)

// ---------------------------------------------------------------------------
// One-time: W_stack [L,D,2D] fp32 -> transposed bf16 hi/lo planes [L,2D,D]
// ---------------------------------------------------------------------------
__global__ void k_prep_w(const float* __restrict__ W,
                         unsigned short* __restrict__ WTh,
                         unsigned short* __restrict__ WTl) {
  int idx = blockIdx.x * blockDim.x + threadIdx.x;  // L*DD*TWOD threads
  int l   = idx / (DD * TWOD);
  int rem = idx - l * DD * TWOD;
  int k   = rem / TWOD;
  int c   = rem - k * TWOD;
  float v = W[idx];
  unsigned short h  = f2bf(v);
  unsigned short lo = f2bf(v - bf2f(h));
  size_t o = ((size_t)l * TWOD + c) * DD + k;
  WTh[o] = h; WTl[o] = lo;
}

// x fp32 -> bf16 hi/lo planes (layer 0 input only; later layers fused in k_pv)
__global__ void k_cvt_x(const float* __restrict__ x,
                        unsigned short* __restrict__ xh,
                        unsigned short* __restrict__ xl) {
  int idx = blockIdx.x * blockDim.x + threadIdx.x;  // NN*DD threads
  float v = x[idx];
  unsigned short h = f2bf(v);
  xh[idx] = h;
  xl[idx] = f2bf(v - bf2f(h));
}

// ---------------------------------------------------------------------------
// proj = x @ W  -> Wh (row-major + transposed) and Ws (row-major), bf16 hi/lo
// One wave computes a 16x64 tile strip; split-bf16: Ah*Bh + Ah*Bl + Al*Bh.
// ---------------------------------------------------------------------------
__global__ void __launch_bounds__(256) k_proj(
    const unsigned short* __restrict__ xh, const unsigned short* __restrict__ xl,
    const unsigned short* __restrict__ WTh, const unsigned short* __restrict__ WTl,
    unsigned short* __restrict__ Whh,  unsigned short* __restrict__ Whl,
    unsigned short* __restrict__ WhTh, unsigned short* __restrict__ WhTl,
    unsigned short* __restrict__ Wsh,  unsigned short* __restrict__ Wsl) {
  int wave = threadIdx.x >> 5, lane = threadIdx.x & 31;
  int strip = blockIdx.x * 8 + wave;     // 8192 strips = 512 m-strips * 16 c-quads
  int m0 = (strip >> 4) * 16;
  int c0 = (strip & 15) * 64;
  int l15 = lane & 15, ksel = lane >> 4;

  v8f acc[4] = {{}, {}, {}, {}};
  const unsigned short* pAh = xh + (size_t)(m0 + l15) * DD + ksel * 8;
  const unsigned short* pAl = xl + (size_t)(m0 + l15) * DD + ksel * 8;

  for (int k0 = 0; k0 < DD; k0 += 32) {
    v16bf Ah = load_frag(pAh + k0);
    v16bf Al = load_frag(pAl + k0);
#pragma unroll
    for (int t = 0; t < 4; ++t) {
      size_t bo = (size_t)(c0 + t * 16 + l15) * DD + k0 + ksel * 8;
      v16bf Bh = load_frag(WTh + bo);
      v16bf Bl = load_frag(WTl + bo);
      acc[t] = WMMA_BF16(Ah, Bh, acc[t]);
      acc[t] = WMMA_BF16(Ah, Bl, acc[t]);
      acc[t] = WMMA_BF16(Al, Bh, acc[t]);
    }
  }

  int mb = (lane >> 4) * 8;
#pragma unroll
  for (int t = 0; t < 4; ++t) {
    int c = c0 + t * 16 + l15;
#pragma unroll
    for (int r = 0; r < 8; ++r) {
      int row = m0 + mb + r;
      float v = acc[t][r];
      unsigned short h  = f2bf(v);
      unsigned short lo = f2bf(v - bf2f(h));
      if (c < DD) {  // Wh half
        size_t o = (size_t)row * DD + c;
        Whh[o] = h; Whl[o] = lo;
        size_t ot = (size_t)c * NN + row;  // transposed copy for PV B-operand
        WhTh[ot] = h; WhTl[ot] = lo;
      } else {       // Ws half
        size_t o = (size_t)row * DD + (c - DD);
        Wsh[o] = h; Wsl[o] = lo;
      }
    }
  }
}

// ---------------------------------------------------------------------------
// S = Wh @ Ws^T, stored fp32; per-row online (max, sum-exp) partials.
// Grid: 512 row-strips x 8 column chunks; wave handles 128 columns (8 tiles).
// ---------------------------------------------------------------------------
__global__ void __launch_bounds__(256) k_scores(
    const unsigned short* __restrict__ Whh, const unsigned short* __restrict__ Whl,
    const unsigned short* __restrict__ Wsh, const unsigned short* __restrict__ Wsl,
    float* __restrict__ S, float* __restrict__ Mpart, float* __restrict__ Lpart) {
  int wave = threadIdx.x >> 5, lane = threadIdx.x & 31;
  int strip = blockIdx.x >> 3, chunk = blockIdx.x & 7;
  int m0 = strip * 16;
  int colBase = chunk * 1024 + wave * 128;
  int l15 = lane & 15, ksel = lane >> 4, mb = (lane >> 4) * 8;

  float mx[8], sm[8];
#pragma unroll
  for (int r = 0; r < 8; ++r) { mx[r] = -3.0e38f; sm[r] = 0.f; }

  const unsigned short* pAh = Whh + (size_t)(m0 + l15) * DD + ksel * 8;
  const unsigned short* pAl = Whl + (size_t)(m0 + l15) * DD + ksel * 8;

  for (int ct = 0; ct < 8; ++ct) {
    int j0 = colBase + ct * 16;
    v8f acc = {};
    const unsigned short* pBh = Wsh + (size_t)(j0 + l15) * DD + ksel * 8;
    const unsigned short* pBl = Wsl + (size_t)(j0 + l15) * DD + ksel * 8;
    for (int k0 = 0; k0 < DD; k0 += 32) {
      v16bf Ah = load_frag(pAh + k0);
      v16bf Al = load_frag(pAl + k0);
      v16bf Bh = load_frag(pBh + k0);
      v16bf Bl = load_frag(pBl + k0);
      acc = WMMA_BF16(Ah, Bh, acc);
      acc = WMMA_BF16(Ah, Bl, acc);
      acc = WMMA_BF16(Al, Bh, acc);
    }
#pragma unroll
    for (int r = 0; r < 8; ++r) {
      int row = m0 + mb + r;
      float s = acc[r];
      S[(size_t)row * NN + j0 + l15] = s;
      float mo = mx[r], mn = fmaxf(mo, s);
      sm[r] = sm[r] * __expf(mo - mn) + __expf(s - mn);
      mx[r] = mn;
    }
  }
  // reduce (max, sumexp) across the 16 lanes holding one row
#pragma unroll
  for (int r = 0; r < 8; ++r) {
#pragma unroll
    for (int d = 1; d < 16; d <<= 1) {
      float mo = __shfl_xor(mx[r], d, 16);
      float so = __shfl_xor(sm[r], d, 16);
      float mn = fmaxf(mx[r], mo);
      sm[r] = sm[r] * __expf(mx[r] - mn) + so * __expf(mo - mn);
      mx[r] = mn;
    }
  }
  if (l15 == 0) {
    int part = chunk * 8 + wave;  // 64 partials per row
#pragma unroll
    for (int r = 0; r < 8; ++r) {
      int row = m0 + mb + r;
      Mpart[(size_t)row * 64 + part] = mx[r];
      Lpart[(size_t)row * 64 + part] = sm[r];
    }
  }
}

__global__ void k_merge(const float* __restrict__ Mpart, const float* __restrict__ Lpart,
                        float* __restrict__ Mrow, float* __restrict__ Linv) {
  int row = blockIdx.x * blockDim.x + threadIdx.x;
  if (row >= NN) return;
  float m = -3.0e38f;
  for (int i = 0; i < 64; ++i) m = fmaxf(m, Mpart[(size_t)row * 64 + i]);
  float l = 0.f;
  for (int i = 0; i < 64; ++i)
    l += Lpart[(size_t)row * 64 + i] * __expf(Mpart[(size_t)row * 64 + i] - m);
  Mrow[row] = m;
  Linv[row] = 1.0f / l;
}

// ---------------------------------------------------------------------------
// out = (softmax(S)*adj) @ Wh + x, leaky-ReLU; optionally emit bf16 hi/lo of
// the activation for the next layer's projection. P tile built in LDS.
// Block: 16 rows x all 512 cols; 8 waves x 64 cols; K loop over 8192 in 32s.
// ---------------------------------------------------------------------------
__global__ void __launch_bounds__(256) k_pv(
    const float* __restrict__ S, const float* __restrict__ adj,
    const unsigned short* __restrict__ WhTh, const unsigned short* __restrict__ WhTl,
    const float* __restrict__ Mrow, const float* __restrict__ Linv,
    const float* __restrict__ xres, float* __restrict__ outp,
    unsigned short* __restrict__ xh, unsigned short* __restrict__ xl, int writeBF) {
  __shared__ unsigned short Ph[16 * 32];
  __shared__ unsigned short Pl[16 * 32];
  __shared__ float sM[16], sLi[16];

  int tid = threadIdx.x, wave = tid >> 5, lane = tid & 31;
  int m0 = blockIdx.x * 16;
  int d0 = wave * 64;
  int l15 = lane & 15, ksel = lane >> 4, mb = (lane >> 4) * 8;

  if (tid < 16) { sM[tid] = Mrow[m0 + tid]; sLi[tid] = Linv[m0 + tid]; }

  v8f acc[4] = {{}, {}, {}, {}};

  int ei = (tid * 2) >> 5;   // P-fill row (0..15)
  int ej = (tid * 2) & 31;   // P-fill col pair base (even)
  const size_t rowbase = (size_t)(m0 + ei) * NN + ej;

  for (int j0 = 0; j0 < NN; j0 += 32) {
    __syncthreads();  // LDS from previous iteration fully consumed; sM visible
    if (j0 + 32 < NN) {
      __builtin_prefetch(S + rowbase + j0 + 32, 0, 0);
      __builtin_prefetch(adj + rowbase + j0 + 32, 0, 0);
    }
    float s0 = S[rowbase + j0],   s1 = S[rowbase + j0 + 1];
    float a0 = adj[rowbase + j0], a1 = adj[rowbase + j0 + 1];
    float p0 = __expf(s0 - sM[ei]) * sLi[ei] * a0;  // post-softmax mask
    float p1 = __expf(s1 - sM[ei]) * sLi[ei] * a1;
    unsigned short h0 = f2bf(p0), h1 = f2bf(p1);
    Ph[ei * 32 + ej]     = h0;
    Ph[ei * 32 + ej + 1] = h1;
    Pl[ei * 32 + ej]     = f2bf(p0 - bf2f(h0));
    Pl[ei * 32 + ej + 1] = f2bf(p1 - bf2f(h1));
    __syncthreads();

    v16bf Ah = load_frag(Ph + l15 * 32 + ksel * 8);
    v16bf Al = load_frag(Pl + l15 * 32 + ksel * 8);
#pragma unroll
    for (int t = 0; t < 4; ++t) {
      size_t bo = (size_t)(d0 + t * 16 + l15) * NN + j0 + ksel * 8;
      v16bf Bh = load_frag(WhTh + bo);
      v16bf Bl = load_frag(WhTl + bo);
      acc[t] = WMMA_BF16(Ah, Bh, acc[t]);
      acc[t] = WMMA_BF16(Ah, Bl, acc[t]);
      acc[t] = WMMA_BF16(Al, Bh, acc[t]);
    }
  }

#pragma unroll
  for (int t = 0; t < 4; ++t) {
    int col = d0 + t * 16 + l15;
#pragma unroll
    for (int r = 0; r < 8; ++r) {
      int row = m0 + mb + r;
      float v = acc[t][r] + xres[(size_t)row * DD + col];  // residual
      v = (v > 0.f) ? v : NEG_SLOPE * v;                   // leaky relu
      outp[(size_t)row * DD + col] = v;
      if (writeBF) {
        unsigned short h = f2bf(v);
        xh[(size_t)row * DD + col] = h;
        xl[(size_t)row * DD + col] = f2bf(v - bf2f(h));
      }
    }
  }
}

// ---------------------------------------------------------------------------
extern "C" void kernel_launch(void* const* d_in, const int* in_sizes, int n_in,
                              void* d_out, int out_size, void* d_ws, size_t ws_size,
                              hipStream_t stream) {
  (void)in_sizes; (void)n_in; (void)out_size; (void)ws_size;
  const float* x_in = (const float*)d_in[0];
  const float* adj  = (const float*)d_in[1];
  const float* Wst  = (const float*)d_in[2];
  float* out = (float*)d_out;

  char* ws = (char*)d_ws;
  size_t off = 0;
  auto alloc = [&](size_t bytes) -> char* {
    char* p = ws + off;
    off += (bytes + 255) & ~(size_t)255;
    return p;
  };
  const size_t ND = (size_t)NN * DD;                        // 4.19M elems
  unsigned short* xh   = (unsigned short*)alloc(ND * 2);
  unsigned short* xl   = (unsigned short*)alloc(ND * 2);
  unsigned short* WTh  = (unsigned short*)alloc((size_t)NLAYER * TWOD * DD * 2);
  unsigned short* WTl  = (unsigned short*)alloc((size_t)NLAYER * TWOD * DD * 2);
  unsigned short* Whh  = (unsigned short*)alloc(ND * 2);
  unsigned short* Whl  = (unsigned short*)alloc(ND * 2);
  unsigned short* WhTh = (unsigned short*)alloc(ND * 2);
  unsigned short* WhTl = (unsigned short*)alloc(ND * 2);
  unsigned short* Wsh  = (unsigned short*)alloc(ND * 2);
  unsigned short* Wsl  = (unsigned short*)alloc(ND * 2);
  float* x1    = (float*)alloc(ND * 4);
  float* Mpart = (float*)alloc((size_t)NN * 64 * 4);
  float* Lpart = (float*)alloc((size_t)NN * 64 * 4);
  float* Mrow  = (float*)alloc((size_t)NN * 4);
  float* Linv  = (float*)alloc((size_t)NN * 4);
  float* Sbuf  = (float*)alloc((size_t)NN * NN * 4);        // 268 MB, last

  k_prep_w<<<(NLAYER * DD * TWOD) / 256, 256, 0, stream>>>(Wst, WTh, WTl);
  k_cvt_x<<<(int)(ND / 256), 256, 0, stream>>>(x_in, xh, xl);

  for (int l = 0; l < NLAYER; ++l) {
    const unsigned short* wth = WTh + (size_t)l * TWOD * DD;
    const unsigned short* wtl = WTl + (size_t)l * TWOD * DD;
    k_proj<<<1024, 256, 0, stream>>>(xh, xl, wth, wtl,
                                     Whh, Whl, WhTh, WhTl, Wsh, Wsl);
    k_scores<<<4096, 256, 0, stream>>>(Whh, Whl, Wsh, Wsl, Sbuf, Mpart, Lpart);
    k_merge<<<NN / 256, 256, 0, stream>>>(Mpart, Lpart, Mrow, Linv);
    const float* xres = (l == 0) ? x_in : x1;
    float* o = (l == NLAYER - 1) ? out : x1;
    k_pv<<<NN / 16, 256, 0, stream>>>(Sbuf, adj, WhTh, WhTl, Mrow, Linv,
                                      xres, o, xh, xl, (l == NLAYER - 1) ? 0 : 1);
  }
}
